// HConstructor_47107201303169
// MI455X (gfx1250) — matrix operand order repaired and checked by reading, not compiled, not run on hardware
//
#include <hip/hip_runtime.h>
#include <hip/hip_bf16.h>
#include <math.h>

typedef __attribute__((ext_vector_type(16))) __bf16 v16bf;
typedef __attribute__((ext_vector_type(8)))  __bf16 v8bf;
typedef __attribute__((ext_vector_type(8)))  float  v8f;

// ---------------------------------------------------------------------------
// gfx1250 async copy helpers (ASYNCcnt-tracked LDS fills).
// LDS operand = low 32 bits of a generic pointer to __shared__ (the LDS
// offset); global operand = 64-bit VGPR pair.
// ---------------------------------------------------------------------------
__device__ __forceinline__ unsigned int lds_off_of(const void* p) {
    return (unsigned int)(size_t)p;
}
__device__ __forceinline__ void async_copy_b128(unsigned int lds_off, const void* g) {
    asm volatile("global_load_async_to_lds_b128 %0, %1, off"
                 :: "v"(lds_off), "v"(g) : "memory");
}
__device__ __forceinline__ void wait_async0() {
    asm volatile("s_wait_asynccnt 0x0" ::: "memory");
}

// ---------------------------------------------------------------------------
// Elementwise: transpose + f32 -> bf16 weight conversion.
// W: [K, Ncols] row-major f32.   Wt: [Ncols, K] row-major bf16.
// ---------------------------------------------------------------------------
__global__ __launch_bounds__(256) void wt_convert_kernel(
    const float* __restrict__ W, __bf16* __restrict__ Wt, int K, int Ncols)
{
    size_t idx = (size_t)blockIdx.x * 256 + threadIdx.x;
    size_t total = (size_t)K * Ncols;
    if (idx >= total) return;
    int kk = (int)(idx / Ncols);
    int n  = (int)(idx % Ncols);
    Wt[(size_t)n * K + kk] = (__bf16)W[idx];
}

// ---------------------------------------------------------------------------
// Row LayerNorm -> bf16 out.  One block per row.
// ---------------------------------------------------------------------------
__global__ __launch_bounds__(256) void layernorm_rows_kernel(
    const float* __restrict__ in, const float* __restrict__ g,
    const float* __restrict__ b, __bf16* __restrict__ out, int ldo, int D)
{
    __shared__ float s1[256], s2[256];
    int row = blockIdx.x, t = threadIdx.x;
    const float* r = in + (size_t)row * D;
    float sum = 0.f, sq = 0.f;
    for (int i = t; i < D; i += 256) { float x = r[i]; sum += x; sq += x * x; }
    s1[t] = sum; s2[t] = sq; __syncthreads();
    for (int o = 128; o > 0; o >>= 1) {
        if (t < o) { s1[t] += s1[t + o]; s2[t] += s2[t + o]; }
        __syncthreads();
    }
    float mean = s1[0] / D;
    float var  = s2[0] / D - mean * mean;
    float inv  = rsqrtf(var + 1e-5f);
    __bf16* orow = out + (size_t)row * ldo;
    for (int i = t; i < D; i += 256)
        orow[i] = (__bf16)((r[i] - mean) * inv * g[i] + b[i]);
}

// ---------------------------------------------------------------------------
// edges = LN(edges_mu + exp(logsigma) * noise) -> bf16 (first half of concat)
// ---------------------------------------------------------------------------
__global__ __launch_bounds__(256) void edges_prep_kernel(
    const float* __restrict__ noise, const float* __restrict__ mu,
    const float* __restrict__ ls, const float* __restrict__ g,
    const float* __restrict__ b, __bf16* __restrict__ out, int ldo, int D)
{
    __shared__ float s1[256], s2[256];
    int row = blockIdx.x, t = threadIdx.x;
    const float* nr = noise + (size_t)row * D;
    float sum = 0.f, sq = 0.f;
    for (int i = t; i < D; i += 256) {
        float e = mu[i] + expf(ls[i]) * nr[i];
        sum += e; sq += e * e;
    }
    s1[t] = sum; s2[t] = sq; __syncthreads();
    for (int o = 128; o > 0; o >>= 1) {
        if (t < o) { s1[t] += s1[t + o]; s2[t] += s2[t + o]; }
        __syncthreads();
    }
    float mean = s1[0] / D;
    float var  = s2[0] / D - mean * mean;
    float inv  = rsqrtf(var + 1e-5f);
    __bf16* orow = out + (size_t)row * ldo;
    for (int i = t; i < D; i += 256) {
        float e = mu[i] + expf(ls[i]) * nr[i];
        orow[i] = (__bf16)((e - mean) * inv * g[i] + b[i]);
    }
}

// ---------------------------------------------------------------------------
// Tiled WMMA bf16 GEMM with async-to-LDS double buffering.
//   C[M,Ncols] = act( (A[M,K] @ Bt[Ncols,K]^T) * scale + bias )
// Workgroup = 256 threads (8 waves) computing a 128x128 C tile.
// Wave (wm,wn) computes a 32x64 sub-tile = 2x4 WMMA 16x16 tiles.
// K staged in 32-wide slabs via global_load_async_to_lds_b128 (ASYNCcnt),
// fragments read back with ds_load_b128, math on v_wmma_f32_16x16x32_bf16.
// Requires M%128==0, Ncols%128==0, K%32==0 (true for all shapes here).
// ---------------------------------------------------------------------------
#define BM 128
#define BN 128
#define BK 32

__global__ __launch_bounds__(256) void gemm_wmma_bf16_kernel(
    const __bf16* __restrict__ A,  int lda,
    const __bf16* __restrict__ Bt, int ldb,
    const float*  __restrict__ bias,
    float*  __restrict__ outF,   // may be null
    __bf16* __restrict__ outB,   // may be null
    int ldc,
    __bf16* __restrict__ outBT,  // may be null (transposed bf16 store)
    int ldtc,
    int K, float scale, int relu)
{
    __shared__ __bf16 smA[2][BM * BK];   // row-major, 32 bf16 (64B) per row
    __shared__ __bf16 smB[2][BN * BK];

    const int t    = threadIdx.x;
    const int w    = t >> 5;            // wave id 0..7
    const int lane = t & 31;
    const int l15  = lane & 15;
    const int half = lane >> 4;
    const int wm   = w & 3;             // M quadrant: rows wm*32 .. +32
    const int wn   = w >> 2;            // N half:    cols wn*64 .. +64

    const __bf16* Ag0 = A  + (size_t)(blockIdx.y * BM) * lda;
    const __bf16* Bg0 = Bt + (size_t)(blockIdx.x * BN) * ldb;

    // Issue one 128x32 slab of A and B into LDS buffer `buf`.
    // 512 16B chunks per operand; each of 256 threads issues 2 per operand.
    auto issue = [&](int kk, int buf) {
        const unsigned int la = lds_off_of(&smA[buf][0]);
        const unsigned int lb = lds_off_of(&smB[buf][0]);
        #pragma unroll
        for (int i = 0; i < 2; ++i) {
            const int c   = t + i * 256;   // chunk id 0..511
            const int row = c >> 2;        // tile row 0..127
            const int j   = c & 3;         // 16B chunk within 64B row
            async_copy_b128(la + row * 64 + j * 16,
                            Ag0 + (size_t)row * lda + kk + j * 8);
            async_copy_b128(lb + row * 64 + j * 16,
                            Bg0 + (size_t)row * ldb + kk + j * 8);
        }
    };

    // ISA fragment layouts (16-bit, 16x32 A / 32x16 B):
    auto frag_a = [&](const __bf16* rp) {   // K = {half*8+e, 16+half*8+e}
        v8bf x0 = *(const v8bf*)(rp + half * 8);
        v8bf x1 = *(const v8bf*)(rp + 16 + half * 8);
        v16bf f;
        #pragma unroll
        for (int e = 0; e < 8; ++e) { f[e] = x0[e]; f[e + 8] = x1[e]; }
        return f;
    };
    auto frag_b = [&](const __bf16* rp) {   // K = {half*8+e, 16+half*8+e}
        v8bf x0 = *(const v8bf*)(rp + half * 8);
        v8bf x1 = *(const v8bf*)(rp + 16 + half * 8);
        v16bf f;
        #pragma unroll
        for (int e = 0; e < 8; ++e) { f[e] = x0[e]; f[e + 8] = x1[e]; }
        return f;
    };

    v8f acc[2][4];
    #pragma unroll
    for (int mi = 0; mi < 2; ++mi)
        #pragma unroll
        for (int ni = 0; ni < 4; ++ni)
            #pragma unroll
            for (int r = 0; r < 8; ++r) acc[mi][ni][r] = 0.f;

    issue(0, 0);
    int buf = 0;
    for (int kk = 0; kk < K; kk += BK, buf ^= 1) {
        wait_async0();          // slab `buf` resident in LDS for this wave
        __syncthreads();        // ...and for every wave; prev compute done
        if (kk + BK < K) issue(kk + BK, buf ^ 1);   // overlap next fill

        v16bf af[2], bfv[4];
        #pragma unroll
        for (int mi = 0; mi < 2; ++mi)
            af[mi] = frag_a(&smA[buf][(wm * 32 + mi * 16 + l15) * BK]);
        #pragma unroll
        for (int ni = 0; ni < 4; ++ni)
            bfv[ni] = frag_b(&smB[buf][(wn * 64 + ni * 16 + l15) * BK]);

        #pragma unroll
        for (int mi = 0; mi < 2; ++mi)
            #pragma unroll
            for (int ni = 0; ni < 4; ++ni)
                acc[mi][ni] = __builtin_amdgcn_wmma_f32_16x16x32_bf16(
                    false, af[mi], false, bfv[ni],
                    (short)0, acc[mi][ni], false, false);
    }

    // Epilogue. C tile layout: VGPR r -> row = half*8 + r, col = lane&15.
    #pragma unroll
    for (int mi = 0; mi < 2; ++mi) {
        #pragma unroll
        for (int ni = 0; ni < 4; ++ni) {
            const int col = blockIdx.x * BN + wn * 64 + ni * 16 + l15;
            const float bval = bias ? bias[col] : 0.f;
            #pragma unroll
            for (int r = 0; r < 8; ++r) {
                const int row = blockIdx.y * BM + wm * 32 + mi * 16 + half * 8 + r;
                float v = acc[mi][ni][r] * scale + bval;
                if (relu) v = fmaxf(v, 0.f);
                if (outF)  outF[(size_t)row * ldc + col] = v;
                if (outB)  outB[(size_t)row * ldc + col] = (__bf16)v;
                if (outBT) outBT[(size_t)col * ldtc + row] = (__bf16)v;
            }
        }
    }
}

// ---------------------------------------------------------------------------
// Row softmax (+optional eps-renorm) + top-k mask via 32-step value bisection.
// One block per row; row cached in LDS (cols <= 8192).
// ---------------------------------------------------------------------------
__global__ __launch_bounds__(256) void softmax_topk_rows_kernel(
    const float* __restrict__ in, int ldi, int cols,
    const int* __restrict__ kptr, int add_eps,
    float* __restrict__ outF, __bf16* __restrict__ outB, int ldo)
{
    __shared__ float srow[8192];
    __shared__ float sf[256];
    __shared__ int   si[256];
    const int row = blockIdx.x, t = threadIdx.x;
    const float* r = in + (size_t)row * ldi;

    float mx = -3.4e38f;
    for (int i = t; i < cols; i += 256) { float x = r[i]; srow[i] = x; mx = fmaxf(mx, x); }
    sf[t] = mx; __syncthreads();
    for (int o = 128; o > 0; o >>= 1) { if (t < o) sf[t] = fmaxf(sf[t], sf[t + o]); __syncthreads(); }
    mx = sf[0]; __syncthreads();

    float sum = 0.f;
    for (int i = t; i < cols; i += 256) { float e = expf(srow[i] - mx); srow[i] = e; sum += e; }
    sf[t] = sum; __syncthreads();
    for (int o = 128; o > 0; o >>= 1) { if (t < o) sf[t] += sf[t + o]; __syncthreads(); }
    sum = sf[0]; __syncthreads();

    const float inv    = 1.f / sum;
    const float renorm = add_eps ? 1.f / (1.f + cols * 1e-8f) : 1.f;
    for (int i = t; i < cols; i += 256) {
        float p = srow[i] * inv;
        if (add_eps) p = (p + 1e-8f) * renorm;
        srow[i] = p;
    }
    __syncthreads();

    const int k = *kptr;
    float lo = -1.f;
    if (k < cols) {
        lo = 0.f; float hi = 1.f;
        for (int it = 0; it < 32; ++it) {
            const float mid = 0.5f * (lo + hi);
            int c = 0;
            for (int i = t; i < cols; i += 256) c += (srow[i] > mid) ? 1 : 0;
            si[t] = c; __syncthreads();
            for (int o = 128; o > 0; o >>= 1) { if (t < o) si[t] += si[t + o]; __syncthreads(); }
            const int cnt = si[0]; __syncthreads();
            if (cnt >= k) lo = mid; else hi = mid;
        }
    }

    for (int i = t; i < cols; i += 256) {
        const float p = srow[i];
        const float m = (p > lo) ? p : 0.f;
        if (outF) outF[(size_t)row * ldo + i] = m;
        if (outB) outB[(size_t)row * ldo + i] = (__bf16)m;
    }
}

// ---------------------------------------------------------------------------
// Host launcher
// ---------------------------------------------------------------------------
extern "C" void kernel_launch(void* const* d_in, const int* in_sizes, int n_in,
                              void* d_out, int out_size, void* d_ws, size_t ws_size,
                              hipStream_t stream)
{
    const float* inputs  = (const float*)d_in[0];
    const float* noise   = (const float*)d_in[1];
    const float* e_mu    = (const float*)d_in[2];
    const float* e_ls    = (const float*)d_in[3];
    const float* Wq      = (const float*)d_in[4];
    const float* bq      = (const float*)d_in[5];
    const float* Wk      = (const float*)d_in[6];
    const float* bk      = (const float*)d_in[7];
    const float* Wv      = (const float*)d_in[8];
    const float* bv      = (const float*)d_in[9];
    const float* W1      = (const float*)d_in[10];
    const float* b1      = (const float*)d_in[11];
    const float* W2      = (const float*)d_in[12];
    const float* b2      = (const float*)d_in[13];
    const float* g_in    = (const float*)d_in[14];
    const float* b_in    = (const float*)d_in[15];
    const float* g_e     = (const float*)d_in[16];
    const float* b_e     = (const float*)d_in[17];
    const int*   k_n     = (const int*)d_in[18];
    const int*   k_e     = (const int*)d_in[19];

    const int D   = in_sizes[2];            // 1024
    const int N   = in_sizes[0] / D;        // 8192
    const int NS  = in_sizes[1] / D;        // 512
    const int HID = in_sizes[11];           // 1024
    const int D2  = in_sizes[10] / HID;     // 2048
    const float scale = 1.0f / sqrtf((float)D);

    // d_out layout: edges [NS,D] | H [N,NS] | dots [NS,N]
    float* edges_out = (float*)d_out;
    float* H_out     = edges_out + (size_t)NS * D;
    float* dots_out  = H_out + (size_t)N * NS;

    // Workspace carve-out (256B aligned)
    char* p = (char*)d_ws;
    auto alloc = [&](size_t bytes) -> char* {
        char* r = p; p += (bytes + 255) & ~(size_t)255; return r;
    };
    __bf16* Wqt   = (__bf16*)alloc((size_t)D * D * 2);
    __bf16* Wkt   = (__bf16*)alloc((size_t)D * D * 2);
    __bf16* Wvt   = (__bf16*)alloc((size_t)D * D * 2);
    __bf16* W1t   = (__bf16*)alloc((size_t)HID * D2 * 2);
    __bf16* W2t   = (__bf16*)alloc((size_t)D * HID * 2);
    __bf16* x_bf  = (__bf16*)alloc((size_t)N * D * 2);
    __bf16* ecat  = (__bf16*)alloc((size_t)NS * D2 * 2);   // [edges_norm | updates]
    __bf16* q_bf  = (__bf16*)alloc((size_t)NS * D * 2);
    __bf16* k_bf  = (__bf16*)alloc((size_t)N * D * 2);
    __bf16* vt_bf = (__bf16*)alloc((size_t)D * N * 2);     // v transposed [D,N]
    __bf16* at_bf = (__bf16*)alloc((size_t)NS * N * 2);    // masked attn
    __bf16* h_bf  = (__bf16*)alloc((size_t)NS * HID * 2);
    __bf16* e2_bf = (__bf16*)alloc((size_t)NS * D * 2);
    __bf16* k2_bf = (__bf16*)alloc((size_t)NS * D * 2);
    __bf16* q2_bf = (__bf16*)alloc((size_t)N * D * 2);
    (void)ws_size; (void)n_in; (void)out_size;

    auto cdiv = [](size_t a, size_t b) { return (unsigned)((a + b - 1) / b); };

    // 1) weight transpose + bf16 convert
    wt_convert_kernel<<<cdiv((size_t)D * D, 256), 256, 0, stream>>>(Wq, Wqt, D, D);
    wt_convert_kernel<<<cdiv((size_t)D * D, 256), 256, 0, stream>>>(Wk, Wkt, D, D);
    wt_convert_kernel<<<cdiv((size_t)D * D, 256), 256, 0, stream>>>(Wv, Wvt, D, D);
    wt_convert_kernel<<<cdiv((size_t)D2 * HID, 256), 256, 0, stream>>>(W1, W1t, D2, HID);
    wt_convert_kernel<<<cdiv((size_t)HID * D, 256), 256, 0, stream>>>(W2, W2t, HID, D);

    // 2) x = LN(inputs) ; edges_norm = LN(mu + exp(ls)*noise)
    layernorm_rows_kernel<<<N, 256, 0, stream>>>(inputs, g_in, b_in, x_bf, D, D);
    edges_prep_kernel<<<NS, 256, 0, stream>>>(noise, e_mu, e_ls, g_e, b_e, ecat, D2, D);

    // 3) k = relu(x@Wk+bk) ; v = relu(x@Wv+bv) (stored transposed)
    gemm_wmma_bf16_kernel<<<dim3(D / 128, N / 128), 256, 0, stream>>>(
        x_bf, D, Wkt, D, bk, nullptr, k_bf, D, nullptr, 0, D, 1.f, 1);
    gemm_wmma_bf16_kernel<<<dim3(D / 128, N / 128), 256, 0, stream>>>(
        x_bf, D, Wvt, D, bv, nullptr, nullptr, D, vt_bf, N, D, 1.f, 1);

    // 4) q = relu(edges_norm@Wq+bq)
    gemm_wmma_bf16_kernel<<<dim3(D / 128, NS / 128), 256, 0, stream>>>(
        ecat, D2, Wqt, D, bq, nullptr, q_bf, D, nullptr, 0, D, 1.f, 1);

    // 5) dots = q @ k^T * scale  -> d_out (dots slice)
    gemm_wmma_bf16_kernel<<<dim3(N / 128, NS / 128), 256, 0, stream>>>(
        q_bf, D, k_bf, D, nullptr, dots_out, nullptr, N, nullptr, 0, D, scale, 0);

    // 6) attn = topk64(renorm(softmax(dots)+eps)) -> bf16
    softmax_topk_rows_kernel<<<NS, 256, 0, stream>>>(
        dots_out, N, N, k_n, 1, nullptr, at_bf, N);

    // 7) updates = attn @ v -> second half of concat buffer
    gemm_wmma_bf16_kernel<<<dim3(D / 128, NS / 128), 256, 0, stream>>>(
        at_bf, N, vt_bf, N, nullptr, nullptr, ecat + D, D2, nullptr, 0, N, 1.f, 0);

    // 8) MLP: h = relu(ecat@W1+b1) ; edges = h@W2+b2 -> d_out (edges slice) + bf16
    gemm_wmma_bf16_kernel<<<dim3(HID / 128, NS / 128), 256, 0, stream>>>(
        ecat, D2, W1t, D2, b1, nullptr, h_bf, HID, nullptr, 0, D2, 1.f, 1);
    gemm_wmma_bf16_kernel<<<dim3(D / 128, NS / 128), 256, 0, stream>>>(
        h_bf, HID, W2t, HID, b2, edges_out, e2_bf, D, nullptr, 0, HID, 1.f, 0);

    // 9) k2 = relu(edges@Wk+bk) ; q2 = x@Wq+bq (no relu)
    gemm_wmma_bf16_kernel<<<dim3(D / 128, NS / 128), 256, 0, stream>>>(
        e2_bf, D, Wkt, D, bk, nullptr, k2_bf, D, nullptr, 0, D, 1.f, 1);
    gemm_wmma_bf16_kernel<<<dim3(D / 128, N / 128), 256, 0, stream>>>(
        x_bf, D, Wqt, D, bq, nullptr, q2_bf, D, nullptr, 0, D, 1.f, 0);

    // 10) dots_v = q2 @ k2^T * scale -> d_out (H slice), then softmax+top-8 in place
    gemm_wmma_bf16_kernel<<<dim3(NS / 128, N / 128), 256, 0, stream>>>(
        q2_bf, D, k2_bf, D, nullptr, H_out, nullptr, NS, nullptr, 0, D, scale, 0);
    softmax_topk_rows_kernel<<<N, 256, 0, stream>>>(
        H_out, NS, NS, k_e, 0, H_out, nullptr, NS);
}
